// CrissCrossAttention_23845658427766
// MI455X (gfx1250) — compile-verified, matrix operand order checked by reading, not run on hardware
//
#include <hip/hip_runtime.h>
#include <hip/hip_bf16.h>

typedef __attribute__((ext_vector_type(16))) _Float16 v16h;
typedef __attribute__((ext_vector_type(8)))  _Float16 v8h;
typedef __attribute__((ext_vector_type(4)))  _Float16 v4h;
typedef __attribute__((ext_vector_type(8)))  float    v8f;
typedef __attribute__((ext_vector_type(4)))  float    v4f;
typedef __attribute__((ext_vector_type(4)))  int      v4i;

#define BB   4
#define CC   512
#define CQ   64
#define HH   128
#define WW   128
#define HWT  16384   // H*W

// ---------------------------------------------------------------------------
// Async global->LDS b128 copy (gfx1250 GLOBAL_LOAD_ASYNC_TO_LDS_B128,
// tracked by ASYNCcnt). Builtin signature (from this toolchain's diagnostic):
//   void (int4 addrspace(1)* src, int4 addrspace(3)* dst, imm offset, imm cpol)
// Guarded: falls back to a synchronous v8h copy if the builtin is missing.
// ---------------------------------------------------------------------------
#if defined(__gfx1250__) && __has_builtin(__builtin_amdgcn_global_load_async_to_lds_b128)
#define CC_HAS_ASYNC_LDS 1
typedef __attribute__((address_space(1))) v4i* gas_v4i_ptr;
typedef __attribute__((address_space(3))) v4i* las_v4i_ptr;
#else
#define CC_HAS_ASYNC_LDS 0
#endif

__device__ __forceinline__ void lds_copy_b128(const _Float16* g, _Float16* l) {
#if CC_HAS_ASYNC_LDS
  __builtin_amdgcn_global_load_async_to_lds_b128((gas_v4i_ptr)g, (las_v4i_ptr)l,
                                                 0, 0);
#else
  *(v8h*)l = *(const v8h*)g;
#endif
}

__device__ __forceinline__ void lds_copy_wait() {
#if CC_HAS_ASYNC_LDS
#if __has_builtin(__builtin_amdgcn_s_wait_asynccnt)
  __builtin_amdgcn_s_wait_asynccnt(0);
#else
  asm volatile("s_wait_asynccnt 0" ::: "memory");
#endif
#endif
}

// ---------------------------------------------------------------------------
// Fragment loader: matrix stored row-major with K contiguous in each row.
// CDNA5 16-bit A/B frag layout (wave32): lane L holds row (L&15);
// kbase = 8*(L>>4); regs 0..3 = K kbase..kbase+7, regs 4..7 = kbase+16..+23.
// Works for LDS or global pointers (generic addressing).
// ---------------------------------------------------------------------------
__device__ __forceinline__ v16h load_frag_rowK(const _Float16* p0, int row,
                                               int rowStride, int k0) {
  const int lane = threadIdx.x & 31;
  const _Float16* p = p0 + (long)(row + (lane & 15)) * rowStride + k0 + ((lane >> 4) << 3);
  v8h lo = *(const v8h*)p;
  v8h hi = *(const v8h*)(p + 16);
  v16h f;
#pragma unroll
  for (int i = 0; i < 8; ++i) { f[i] = lo[i]; f[i + 8] = hi[i]; }
  return f;
}

__device__ __forceinline__ v8f wmma16(v16h a, v16h b, v8f c) {
  return __builtin_amdgcn_wmma_f32_16x16x32_f16(false, a, false, b, (short)0, c,
                                                false, false);
}

// ---------------------------------------------------------------------------
// K0: f32 -> f16 conversion (n multiple of 4)
// ---------------------------------------------------------------------------
__global__ __launch_bounds__(256) void cvt_f32_f16(const float* __restrict__ in,
                                                   _Float16* __restrict__ out,
                                                   long n) {
  long i = ((long)blockIdx.x * blockDim.x + threadIdx.x) * 4;
  if (i >= n) return;
  v4f v = *(const v4f*)(in + i);
  v4h o;
#pragma unroll
  for (int e = 0; e < 4; ++e) o[e] = (_Float16)v[e];
  *(v4h*)(out + i) = o;
}

// ---------------------------------------------------------------------------
// K1: y[b,o,p] = sum_c W[o,c] * x[b,c,p] + bias[o]   (WMMA f16, f32 accum)
// grid: (HW/64, O/64, B), block 128 (4 waves). Each wave: 1 M-subtile x 4 N.
// A-tile staged via async global->LDS b128; B-tile transposed via VGPRs.
// ---------------------------------------------------------------------------
__global__ __launch_bounds__(128) void qkv_gemm(const _Float16* __restrict__ xh,
                                                const _Float16* __restrict__ Wh,
                                                const float* __restrict__ bias,
                                                _Float16* __restrict__ yh, int O) {
  __shared__ __align__(16) _Float16 lA[64 * 32];   // [m][k]
  __shared__ __align__(16) _Float16 lBT[64 * 40];  // [n][k] padded stride 40
  const int b = blockIdx.z;
  const int nBase = blockIdx.x * 64;
  const int mBase = blockIdx.y * 64;
  const int tid = threadIdx.x, lane = tid & 31, wave = tid >> 5;
  const _Float16* xb = xh + (long)b * CC * HWT;
  v8f acc[4] = {};
  for (int k0 = 0; k0 < CC; k0 += 32) {
    // stage A (64x32) row-major, async copy engine
#pragma unroll
    for (int r = 0; r < 2; ++r) {
      int e = tid + 128 * r, row = e >> 2, kc = (e & 3) * 8;
      lds_copy_b128(Wh + (long)(mBase + row) * CC + k0 + kc,
                    lA + row * 32 + kc);
    }
    // stage B (32x64) transposed to [n][k]
#pragma unroll
    for (int r = 0; r < 2; ++r) {
      int e = tid + 128 * r, k = e >> 3, nc = (e & 7) * 8;
      const _Float16* gp = xb + (long)(k0 + k) * HWT + nBase + nc;
      if (k0 + 32 < CC) __builtin_prefetch(gp + 32l * HWT, 0, 1);
      v8h v = *(const v8h*)gp;
#pragma unroll
      for (int j = 0; j < 8; ++j) lBT[(nc + j) * 40 + k] = v[j];
    }
    lds_copy_wait();
    __syncthreads();
    v16h a = load_frag_rowK(lA, wave * 16, 32, 0);
#pragma unroll
    for (int nt = 0; nt < 4; ++nt) {
      v16h bf = load_frag_rowK(lBT, nt * 16, 40, 0);
      acc[nt] = wmma16(a, bf, acc[nt]);
    }
    __syncthreads();
  }
  float bv[8];
#pragma unroll
  for (int v = 0; v < 8; ++v)
    bv[v] = bias[mBase + wave * 16 + v + ((lane >> 4) << 3)];
#pragma unroll
  for (int nt = 0; nt < 4; ++nt) {
    int n = nBase + nt * 16 + (lane & 15);
#pragma unroll
    for (int v = 0; v < 8; ++v) {
      int m = mBase + wave * 16 + v + ((lane >> 4) << 3);
      yh[((long)b * O + m) * HWT + n] = (_Float16)(acc[nt][v] + bv[v]);
    }
  }
}

// ---------------------------------------------------------------------------
// K2: energy_H[b,w,h,i] = sum_c q[b,c,h,w]*k[b,c,i,w].  grid (W, B), 256 thr.
// ---------------------------------------------------------------------------
__global__ __launch_bounds__(256) void energy_h(const _Float16* __restrict__ qh,
                                                const _Float16* __restrict__ kh,
                                                float* __restrict__ eH) {
  __shared__ __align__(16) _Float16 qT[128 * 64];  // [h][c]
  __shared__ __align__(16) _Float16 kT[128 * 64];  // [i][c]
  const int w = blockIdx.x, b = blockIdx.y;
  const int tid = threadIdx.x, lane = tid & 31, wave = tid >> 5;
  {
    int h = tid >> 1, c0 = (tid & 1) * 32;
    const _Float16* qp = qh + (((long)b * CQ + c0) * HH + h) * WW + w;
    const _Float16* kp = kh + (((long)b * CQ + c0) * HH + h) * WW + w;
#pragma unroll 8
    for (int c = 0; c < 32; ++c) {
      qT[h * 64 + c0 + c] = qp[(long)c * HWT];
      kT[h * 64 + c0 + c] = kp[(long)c * HWT];
    }
  }
  __syncthreads();
  v8f acc[8] = {};
#pragma unroll
  for (int kt = 0; kt < 2; ++kt) {
    v16h a = load_frag_rowK(qT, wave * 16, 64, kt * 32);
#pragma unroll
    for (int nt = 0; nt < 8; ++nt) {
      v16h bf = load_frag_rowK(kT, nt * 16, 64, kt * 32);
      acc[nt] = wmma16(a, bf, acc[nt]);
    }
  }
  float* outp = eH + ((long)b * WW + w) * HH * HH;
#pragma unroll
  for (int nt = 0; nt < 8; ++nt) {
    int i = nt * 16 + (lane & 15);
#pragma unroll
    for (int v = 0; v < 8; ++v) {
      int h = wave * 16 + v + ((lane >> 4) << 3);
      outp[(long)h * HH + i] = acc[nt][v];
    }
  }
}

// ---------------------------------------------------------------------------
// K3: energy_W[b,h,w,j] = sum_c q[b,c,h,w]*k[b,c,h,j].  grid (H, B), 256 thr.
// ---------------------------------------------------------------------------
__global__ __launch_bounds__(256) void energy_w(const _Float16* __restrict__ qh,
                                                const _Float16* __restrict__ kh,
                                                float* __restrict__ eW) {
  __shared__ __align__(16) _Float16 qT[128 * 64];  // [w][c]
  __shared__ __align__(16) _Float16 kT[128 * 64];  // [j][c]
  const int h = blockIdx.x, b = blockIdx.y;
  const int tid = threadIdx.x, lane = tid & 31, wave = tid >> 5;
  {
    int c = tid >> 2, w0 = (tid & 3) * 32;
    const _Float16* qp = qh + (((long)b * CQ + c) * HH + h) * WW + w0;
    const _Float16* kp = kh + (((long)b * CQ + c) * HH + h) * WW + w0;
#pragma unroll
    for (int jj = 0; jj < 4; ++jj) {
      v8h qv = *(const v8h*)(qp + jj * 8);
      v8h kv = *(const v8h*)(kp + jj * 8);
#pragma unroll
      for (int e = 0; e < 8; ++e) {
        qT[(w0 + jj * 8 + e) * 64 + c] = qv[e];
        kT[(w0 + jj * 8 + e) * 64 + c] = kv[e];
      }
    }
  }
  __syncthreads();
  v8f acc[8] = {};
#pragma unroll
  for (int kt = 0; kt < 2; ++kt) {
    v16h a = load_frag_rowK(qT, wave * 16, 64, kt * 32);
#pragma unroll
    for (int nt = 0; nt < 8; ++nt) {
      v16h bf = load_frag_rowK(kT, nt * 16, 64, kt * 32);
      acc[nt] = wmma16(a, bf, acc[nt]);
    }
  }
  float* outp = eW + ((long)b * HH + h) * WW * WW;
#pragma unroll
  for (int nt = 0; nt < 8; ++nt) {
    int j = nt * 16 + (lane & 15);
#pragma unroll
    for (int v = 0; v < 8; ++v) {
      int wr = wave * 16 + v + ((lane >> 4) << 3);
      outp[(long)wr * WW + j] = acc[nt][v];
    }
  }
}

// ---------------------------------------------------------------------------
// K4: joint softmax over [energy_H row | energy_W row] with h==i masked.
// One wave per pixel; 8 values/lane; shfl reductions. Writes f16 attn.
// ---------------------------------------------------------------------------
__global__ __launch_bounds__(256) void softmax_cc(const float* __restrict__ eH,
                                                  const float* __restrict__ eW,
                                                  _Float16* __restrict__ aH,
                                                  _Float16* __restrict__ aW) {
  const int tid = threadIdx.x, lane = tid & 31, wave = tid >> 5;
  long p = (long)blockIdx.x * 8 + wave;
  int b = (int)(p >> 14);
  int rem = (int)(p & 16383);
  int h = rem >> 7, w = rem & 127;
  const float* rH = eH + (((long)b * WW + w) * HH + h) * HH;
  const float* rW = eW + (((long)b * HH + h) * WW + w) * WW;
  v4f tH = *(const v4f*)(rH + lane * 4);
  v4f tW = *(const v4f*)(rW + lane * 4);
  float vH[4], vW[4];
  float mx = -__builtin_inff();
#pragma unroll
  for (int e = 0; e < 4; ++e) {
    vH[e] = (lane * 4 + e == h) ? -__builtin_inff() : tH[e];
    vW[e] = tW[e];
    mx = fmaxf(mx, fmaxf(vH[e], vW[e]));
  }
#pragma unroll
  for (int s = 16; s; s >>= 1) mx = fmaxf(mx, __shfl_xor(mx, s, 32));
  float sum = 0.f;
  const float L2E = 1.4426950408889634f;
#pragma unroll
  for (int e = 0; e < 4; ++e) {
    vH[e] = exp2f((vH[e] - mx) * L2E);
    vW[e] = exp2f((vW[e] - mx) * L2E);
    sum += vH[e] + vW[e];
  }
#pragma unroll
  for (int s = 16; s; s >>= 1) sum += __shfl_xor(sum, s, 32);
  float inv = 1.0f / sum;
  v4h oH, oW;
#pragma unroll
  for (int e = 0; e < 4; ++e) {
    oH[e] = (_Float16)(vH[e] * inv);
    oW[e] = (_Float16)(vW[e] * inv);
  }
  *(v4h*)(aH + (((long)b * WW + w) * HH + h) * HH + lane * 4) = oH;
  *(v4h*)(aW + (((long)b * HH + h) * WW + w) * WW + lane * 4) = oW;
}

// ---------------------------------------------------------------------------
// K5: outT[b,c,w,h] = sum_i v[b,c,i,w] * aH[b,w,h,i].  grid (W, B), 256 thr.
// B-fragments stream straight from global (aH is [h][i], i contiguous).
// ---------------------------------------------------------------------------
__global__ __launch_bounds__(256) void agg_h(const _Float16* __restrict__ vh,
                                             const _Float16* __restrict__ aH,
                                             _Float16* __restrict__ outT) {
  __shared__ __align__(16) _Float16 vA[16 * 128];  // [c][i]
  const int w = blockIdx.x, b = blockIdx.y;
  const int tid = threadIdx.x, lane = tid & 31, wave = tid >> 5;
  const _Float16* aBase = aH + ((long)b * WW + w) * HH * HH;
  for (int ct = 0; ct < 32; ++ct) {
    {
      int c = tid >> 4, i0 = (tid & 15) * 8;
      const _Float16* vp =
          vh + (((long)b * CC + ct * 16 + c) * HH + i0) * WW + w;
#pragma unroll
      for (int e = 0; e < 8; ++e) vA[c * 128 + i0 + e] = vp[(long)e * WW];
    }
    __syncthreads();
    v8f acc = {};
#pragma unroll
    for (int kt = 0; kt < 4; ++kt) {
      v16h a = load_frag_rowK(vA, 0, 128, kt * 32);
      v16h bf = load_frag_rowK(aBase, wave * 16, 128, kt * 32);
      acc = wmma16(a, bf, acc);
    }
    int hcol = wave * 16 + (lane & 15);
#pragma unroll
    for (int v = 0; v < 8; ++v) {
      int c = ct * 16 + v + ((lane >> 4) << 3);
      outT[(((long)b * CC + c) * WW + w) * HH + hcol] = (_Float16)acc[v];
    }
    __syncthreads();
  }
}

// ---------------------------------------------------------------------------
// K6: out[b,c,h,w] = sum_j v[b,c,h,j] * aW[b,h,w,j].  grid (H, B), 256 thr.
// V-row tile is a straight b128 copy -> async global->LDS path.
// ---------------------------------------------------------------------------
__global__ __launch_bounds__(256) void agg_w(const _Float16* __restrict__ vh,
                                             const _Float16* __restrict__ aW,
                                             float* __restrict__ out2) {
  __shared__ __align__(16) _Float16 vA[16 * 128];  // [c][j]
  const int h = blockIdx.x, b = blockIdx.y;
  const int tid = threadIdx.x, lane = tid & 31, wave = tid >> 5;
  const _Float16* aBase = aW + ((long)b * HH + h) * WW * WW;
  for (int ct = 0; ct < 32; ++ct) {
    {
      int c = tid >> 4, j0 = (tid & 15) * 8;
      lds_copy_b128(vh + (((long)b * CC + ct * 16 + c) * HH + h) * WW + j0,
                    vA + c * 128 + j0);
    }
    lds_copy_wait();
    __syncthreads();
    v8f acc = {};
#pragma unroll
    for (int kt = 0; kt < 4; ++kt) {
      v16h a = load_frag_rowK(vA, 0, 128, kt * 32);
      v16h bf = load_frag_rowK(aBase, wave * 16, 128, kt * 32);
      acc = wmma16(a, bf, acc);
    }
    int wcol = wave * 16 + (lane & 15);
#pragma unroll
    for (int v = 0; v < 8; ++v) {
      int c = ct * 16 + v + ((lane >> 4) << 3);
      out2[(((long)b * CC + c) * HH + h) * WW + wcol] = acc[v];
    }
    __syncthreads();
  }
}

// ---------------------------------------------------------------------------
// K7: out = gamma*(out + outT^T) + x, LDS-tiled transpose of outT (f16).
// grid: (16 tiles of 32x32, B*C), 256 thr.
// ---------------------------------------------------------------------------
__global__ __launch_bounds__(256) void finalize_cc(const float* __restrict__ x,
                                                   const _Float16* __restrict__ outT,
                                                   const float* __restrict__ gamma,
                                                   float* __restrict__ out) {
  __shared__ float tile[32][33];
  const int bc = blockIdx.y;
  const int t = blockIdx.x;
  const int h0 = (t >> 2) * 32, w0 = (t & 3) * 32;
  const int tid = threadIdx.x;
  const float g = gamma[0];
  {
    int wl = tid >> 3, hl = (tid & 7) * 4;
    const _Float16* p = outT + ((long)bc * WW + (w0 + wl)) * HH + h0 + hl;
    v4h v = *(const v4h*)p;
#pragma unroll
    for (int e = 0; e < 4; ++e) tile[hl + e][wl] = (float)v[e];
  }
  __syncthreads();
  {
    int hl = tid >> 3, wl = (tid & 7) * 4;
    long idx = ((long)bc * HH + h0 + hl) * WW + w0 + wl;
    v4f o = *(const v4f*)(out + idx);
    v4f xv = *(const v4f*)(x + idx);
    v4f r;
#pragma unroll
    for (int e = 0; e < 4; ++e) r[e] = g * (o[e] + tile[hl][wl + e]) + xv[e];
    *(v4f*)(out + idx) = r;
  }
}

// ---------------------------------------------------------------------------
extern "C" void kernel_launch(void* const* d_in, const int* in_sizes, int n_in,
                              void* d_out, int out_size, void* d_ws,
                              size_t ws_size, hipStream_t stream) {
  (void)in_sizes; (void)n_in; (void)out_size; (void)ws_size;
  const float* x     = (const float*)d_in[0];
  const float* Wq    = (const float*)d_in[1];
  const float* bq    = (const float*)d_in[2];
  const float* Wk    = (const float*)d_in[3];
  const float* bk    = (const float*)d_in[4];
  const float* Wv    = (const float*)d_in[5];
  const float* bv    = (const float*)d_in[6];
  const float* gamma = (const float*)d_in[7];
  float* out = (float*)d_out;

  char* ws = (char*)d_ws;
  const size_t MB = 1024ull * 1024ull;
  _Float16* xh   = (_Float16*)(ws + 0);        // 64 MB (x in f16)
  _Float16* outT = (_Float16*)(ws + 0);        // reused after K1
  _Float16* qh   = (_Float16*)(ws + 64 * MB);  // 8 MB
  _Float16* kh   = (_Float16*)(ws + 72 * MB);  // 8 MB
  _Float16* vh   = (_Float16*)(ws + 80 * MB);  // 64 MB
  float*    eH   = (float*)(ws + 144 * MB);    // 32 MB
  float*    eW   = (float*)(ws + 176 * MB);    // 32 MB
  _Float16* aH   = (_Float16*)(ws + 208 * MB); // 16 MB
  _Float16* aW   = (_Float16*)(ws + 224 * MB); // 16 MB
  _Float16* wqh  = (_Float16*)(ws + 240 * MB);            // 64 KB
  _Float16* wkh  = (_Float16*)(ws + 240 * MB + 65536);    // 64 KB
  _Float16* wvh  = (_Float16*)(ws + 240 * MB + 131072);   // 512 KB

  const long nx = (long)BB * CC * HWT;  // 33,554,432
  cvt_f32_f16<<<(unsigned)(nx / 4 / 256), 256, 0, stream>>>(x, xh, nx);
  cvt_f32_f16<<<32, 256, 0, stream>>>(Wq, wqh, (long)CQ * CC);
  cvt_f32_f16<<<32, 256, 0, stream>>>(Wk, wkh, (long)CQ * CC);
  cvt_f32_f16<<<256, 256, 0, stream>>>(Wv, wvh, (long)CC * CC);

  qkv_gemm<<<dim3(HWT / 64, CQ / 64, BB), 128, 0, stream>>>(xh, wqh, bq, qh, CQ);
  qkv_gemm<<<dim3(HWT / 64, CQ / 64, BB), 128, 0, stream>>>(xh, wkh, bk, kh, CQ);
  qkv_gemm<<<dim3(HWT / 64, CC / 64, BB), 128, 0, stream>>>(xh, wvh, bv, vh, CC);

  energy_h<<<dim3(WW, BB), 256, 0, stream>>>(qh, kh, eH);
  energy_w<<<dim3(HH, BB), 256, 0, stream>>>(qh, kh, eW);

  softmax_cc<<<(unsigned)((long)BB * HH * WW / 8), 256, 0, stream>>>(eH, eW, aH, aW);

  agg_h<<<dim3(WW, BB), 256, 0, stream>>>(vh, aH, outT);
  agg_w<<<dim3(HH, BB), 256, 0, stream>>>(vh, aW, out);

  finalize_cc<<<dim3(16, BB * CC), 256, 0, stream>>>(x, outT, gamma, out);
}
